// PositionRelationEncodeUnit_GCN_67731634258427
// MI455X (gfx1250) — matrix operand order repaired and checked by reading, not compiled, not run on hardware
//
#include <hip/hip_runtime.h>
#include <hip/hip_bf16.h>
#include <stdint.h>

// N=512, D_OBJ=256, D_BOX=128, D_IN=640, HID=256, D_OUT=256
#define NN   512
#define DOBJ 256
#define DBOX 128
#define HID  256
#define DOUT 256
#define KSPLIT 4            // j-dimension split for the aggregation pass

typedef __attribute__((ext_vector_type(16))) __bf16 v16bf;
typedef __attribute__((ext_vector_type(8)))  float  v8f;

union FragBF { uint4 q[2]; v16bf v; };

__device__ __forceinline__ unsigned short f2bf(float f) {
  unsigned int u = __float_as_uint(f);
  u += 0x7fffu + ((u >> 16) & 1u);   // round-to-nearest-even
  return (unsigned short)(u >> 16);
}
__device__ __forceinline__ unsigned int pkbf(float x, float y) {
  return (unsigned int)f2bf(x) | ((unsigned int)f2bf(y) << 16);
}
__device__ __forceinline__ v8f wmma_bf16(v16bf a, v16bf b, v8f c) {
  // D(16x16 f32) = A(16x32 bf16) x B(32x16 bf16) + C
  return __builtin_amdgcn_wmma_f32_16x16x32_bf16(false, a, false, b, (short)0, c,
                                                 false, false);
}

// ---------------------------------------------------------------------------
// k0: per-object projections.  A1[i,h]=obj[i]@w1[:, :256].T + b1,
// B1[i,h]=obj[i]@w1[:,256:512].T, same with cw1/cb1 for the conv MLP.
// ---------------------------------------------------------------------------
__global__ void k0_proj(const float* __restrict__ obj,
                        const float* __restrict__ w1, const float* __restrict__ b1,
                        const float* __restrict__ cw1, const float* __restrict__ cb1,
                        float* __restrict__ A1, float* __restrict__ B1,
                        float* __restrict__ A1c, float* __restrict__ B1c) {
  __shared__ float sObj[DOBJ];
  const int i = blockIdx.x, h = threadIdx.x;
  sObj[h] = obj[(size_t)i * DOBJ + h];
  __syncthreads();
  const float* wr  = w1  + (size_t)h * 640;
  const float* cwr = cw1 + (size_t)h * 640;
  float a = 0.f, bq = 0.f, ac = 0.f, bc = 0.f;
#pragma unroll 4
  for (int k = 0; k < DOBJ; ++k) {
    const float o = sObj[k];
    a  += o * wr[k];
    bq += o * wr[DOBJ + k];
    ac += o * cwr[k];
    bc += o * cwr[DOBJ + k];
  }
  const size_t idx = (size_t)i * HID + h;
  A1[idx]  = a + b1[h];
  B1[idx]  = bq;
  A1c[idx] = ac + cb1[h];
  B1c[idx] = bc;
}

// ---------------------------------------------------------------------------
// k_init: fold BatchNorm into per-channel affine for the conf head.
// conf = sum_h w2s[h]*leaky(pre[h]) + Cconst
// ---------------------------------------------------------------------------
__global__ void k_init(const float* __restrict__ w2, const float* __restrict__ b2,
                       const float* __restrict__ gamma, const float* __restrict__ beta,
                       const float* __restrict__ mean, const float* __restrict__ var,
                       float* __restrict__ w2s, float* __restrict__ Cconst) {
  __shared__ float red[HID];
  const int t = threadIdx.x;
  const float s  = gamma[t] * rsqrtf(var[t] + 1e-5f);
  w2s[t] = w2[t] * s;
  const float tt = beta[t] - mean[t] * s;
  red[t] = w2[t] * tt;
  __syncthreads();
  for (int st = HID / 2; st > 0; st >>= 1) {
    if (t < st) red[t] += red[t + st];
    __syncthreads();
  }
  if (t == 0) Cconst[0] = b2[0] + red[0];
}

// ---------------------------------------------------------------------------
// k_cvt_w: bf16 copies of the bbox slices of w1/cw1 (columns 512..639).
// ---------------------------------------------------------------------------
__global__ void k_cvt_w(const float* __restrict__ w1, const float* __restrict__ cw1,
                        unsigned short* __restrict__ wbf,
                        unsigned short* __restrict__ cwbf) {
  const int e = blockIdx.x * blockDim.x + threadIdx.x;   // 0..65535
  const int half = e >> 15;
  const int r = e & 32767;
  const int h = r >> 7, k = r & 127;
  const float* src = half ? cw1 : w1;
  const unsigned short v = f2bf(src[(size_t)h * 640 + 512 + k]);
  if (half) cwbf[r] = v; else wbf[r] = v;
}

// ---------------------------------------------------------------------------
// k1: conf logits.  Block = 16 pairs (row i, cols j0..j0+15); 8 waves cover
// HID=256 (32 columns each, 2 WMMA tiles).  u = bbox_tile(16x128) @ w1c.T.
// ---------------------------------------------------------------------------
__global__ void __launch_bounds__(256) k1_logits(
    const float* __restrict__ bbox,
    const float* __restrict__ A1, const float* __restrict__ B1,
    const unsigned short* __restrict__ wbf,
    const float* __restrict__ w2s, const float* __restrict__ Cconst,
    float* __restrict__ logits) {
  __shared__ unsigned short sB[16 * DBOX];   // bbox tile as bf16
  __shared__ float sRed[8 * 16];
  const int tid = threadIdx.x, lane = tid & 31, wv = tid >> 5;
  const int blk = blockIdx.x;
  const int i  = blk >> 5;
  const int j0 = (blk & 31) << 4;

  // Stage 16x128 f32 (contiguous 8KB) -> bf16 LDS
  const float4* src = (const float4*)(bbox + ((size_t)i * NN + j0) * DBOX);
  const float4 f0 = src[tid * 2], f1 = src[tid * 2 + 1];
  unsigned int* su = (unsigned int*)sB;
  su[tid * 4 + 0] = pkbf(f0.x, f0.y);
  su[tid * 4 + 1] = pkbf(f0.z, f0.w);
  su[tid * 4 + 2] = pkbf(f1.x, f1.y);
  su[tid * 4 + 3] = pkbf(f1.z, f1.w);
  __syncthreads();

  const int hb = wv << 5;          // wave's HID base
  const int ln = lane & 15;
  const int hi = lane >> 4;        // lane half
  const int kA = hi << 4;          // A frag: +16 K for upper lanes
  const int kB = hi << 4;          // B frag: +16 K for upper lanes
  const int mb = hi << 3;          // C rows: M = mb + r
  const uint4* sq = (const uint4*)sB;
  const uint4* wq = (const uint4*)wbf;

  v8f acc0 = {}; v8f acc1 = {};
#pragma unroll
  for (int kc = 0; kc < 4; ++kc) {
    const int koff = kc << 5;
    FragBF A, Bf0, Bf1;
    // A (16x32 bf16): lane ln holds M=ln; K runs [kA, kA+16)
    A.q[0] = sq[ln * 16 + ((koff + kA) >> 3)];
    A.q[1] = sq[ln * 16 + ((koff + kA + 8) >> 3)];
    // B (32x16 bf16): N = ln (+tile*16), K = kB + 0..15
    const int kb = (koff + kB) >> 3;
    Bf0.q[0] = wq[(hb + ln) * 16 + kb];
    Bf0.q[1] = wq[(hb + ln) * 16 + kb + 1];
    Bf1.q[0] = wq[(hb + 16 + ln) * 16 + kb];
    Bf1.q[1] = wq[(hb + 16 + ln) * 16 + kb + 1];
    acc0 = wmma_bf16(A.v, Bf0.v, acc0);
    acc1 = wmma_bf16(A.v, Bf1.v, acc1);
  }

  // Epilogue: per-pair conf partials over this wave's 32 HID columns.
  float v[8];
#pragma unroll
  for (int r = 0; r < 8; ++r) v[r] = 0.f;
#pragma unroll
  for (int t = 0; t < 2; ++t) {
    const int h = hb + (t << 4) + ln;
    const float a1  = A1[(size_t)i * HID + h];
    const float wsc = w2s[h];
#pragma unroll
    for (int r = 0; r < 8; ++r) {
      const float pre = (t ? acc1[r] : acc0[r]) + a1 +
                        B1[(size_t)(j0 + mb + r) * HID + h];
      const float lk = pre > 0.f ? pre : 0.01f * pre;
      v[r] += wsc * lk;
    }
  }
  // reduce over the 16 lanes of each half (different h, same pair rows)
#pragma unroll
  for (int r = 0; r < 8; ++r) {
    v[r] += __shfl_xor(v[r], 1, 32);
    v[r] += __shfl_xor(v[r], 2, 32);
    v[r] += __shfl_xor(v[r], 4, 32);
    v[r] += __shfl_xor(v[r], 8, 32);
  }
  if (ln == 0) {
#pragma unroll
    for (int r = 0; r < 8; ++r) sRed[wv * 16 + mb + r] = v[r];
  }
  __syncthreads();
  if (tid < 16) {
    float s = Cconst[0];
#pragma unroll
    for (int w = 0; w < 8; ++w) s += sRed[w * 16 + tid];
    logits[(size_t)i * NN + j0 + tid] = s;
  }
}

// ---------------------------------------------------------------------------
// k_softmax: row softmax, mask, renormalize (in place); exact rowsum kept.
// ---------------------------------------------------------------------------
__global__ void k_softmax(float* __restrict__ confbuf,
                          const float* __restrict__ mask,
                          float* __restrict__ rowsum) {
  __shared__ float red[256];
  const int i = blockIdx.x, tid = threadIdx.x;
  const float l0 = confbuf[(size_t)i * NN + tid];
  const float l1 = confbuf[(size_t)i * NN + 256 + tid];
  red[tid] = fmaxf(l0, l1); __syncthreads();
  for (int s = 128; s > 0; s >>= 1) { if (tid < s) red[tid] = fmaxf(red[tid], red[tid + s]); __syncthreads(); }
  const float M = red[0]; __syncthreads();
  const float e0 = __expf(l0 - M), e1 = __expf(l1 - M);
  red[tid] = e0 + e1; __syncthreads();
  for (int s = 128; s > 0; s >>= 1) { if (tid < s) red[tid] += red[tid + s]; __syncthreads(); }
  const float invE = 1.f / red[0]; __syncthreads();
  const float p0 = e0 * invE * mask[(size_t)i * NN + tid];
  const float p1 = e1 * invE * mask[(size_t)i * NN + 256 + tid];
  red[tid] = p0 + p1; __syncthreads();
  for (int s = 128; s > 0; s >>= 1) { if (tid < s) red[tid] += red[tid + s]; __syncthreads(); }
  const float S = red[0];
  const float inv = 1.f / (S + 1e-8f);
  confbuf[(size_t)i * NN + tid]       = p0 * inv;
  confbuf[(size_t)i * NN + 256 + tid] = p1 * inv;
  if (tid == 0) rowsum[i] = S * inv;
}

// ---------------------------------------------------------------------------
// k2: Gpart[seg][i,:] = sum_{j in seg} conf[i,j]*relu(A1c[i]+B1c[j]+bbox@cw1c.T)
// Grid = NN*KSPLIT blocks; weight (B) fragments hoisted out of the j loop.
// ---------------------------------------------------------------------------
__global__ void __launch_bounds__(256) k2_aggregate(
    const float* __restrict__ bbox,
    const float* __restrict__ A1c, const float* __restrict__ B1c,
    const unsigned short* __restrict__ cwbf,
    const float* __restrict__ conf,
    float* __restrict__ Gpart) {
  __shared__ unsigned short sB[16 * DBOX];
  __shared__ float sConf[16];
  const int tid = threadIdx.x, lane = tid & 31, wv = tid >> 5;
  const int i   = blockIdx.x >> 2;          // row
  const int seg = blockIdx.x & (KSPLIT - 1);
  const int jlo = seg * (NN / KSPLIT);
  const int jhi = jlo + (NN / KSPLIT);
  const int hb = wv << 5;
  const int ln = lane & 15;
  const int hi = lane >> 4;
  const int kA = hi << 4, kB = hi << 4, mb = hi << 3;
  const float a1c0 = A1c[(size_t)i * HID + hb + ln];
  const float a1c1 = A1c[(size_t)i * HID + hb + 16 + ln];
  const uint4* sq = (const uint4*)sB;
  const uint4* wq = (const uint4*)cwbf;

  // Hoist all 8 weight fragments (2 N-tiles x 4 K-chunks): loop-invariant.
  FragBF Bw[4][2];
#pragma unroll
  for (int kc = 0; kc < 4; ++kc) {
    const int kb = ((kc << 5) + kB) >> 3;
    Bw[kc][0].q[0] = wq[(hb + ln) * 16 + kb];
    Bw[kc][0].q[1] = wq[(hb + ln) * 16 + kb + 1];
    Bw[kc][1].q[0] = wq[(hb + 16 + ln) * 16 + kb];
    Bw[kc][1].q[1] = wq[(hb + 16 + ln) * 16 + kb + 1];
  }

  float g0 = 0.f, g1 = 0.f;
  for (int j0 = jlo; j0 < jhi; j0 += 16) {
    const float4* src = (const float4*)(bbox + ((size_t)i * NN + j0) * DBOX);
    const float4 f0 = src[tid * 2], f1 = src[tid * 2 + 1];
    const float cf = (tid < 16) ? conf[(size_t)i * NN + j0 + tid] : 0.f;
    __syncthreads();                       // previous chunk done with sB/sConf
    unsigned int* su = (unsigned int*)sB;
    su[tid * 4 + 0] = pkbf(f0.x, f0.y);
    su[tid * 4 + 1] = pkbf(f0.z, f0.w);
    su[tid * 4 + 2] = pkbf(f1.x, f1.y);
    su[tid * 4 + 3] = pkbf(f1.z, f1.w);
    if (tid < 16) sConf[tid] = cf;
    __syncthreads();

    v8f acc0 = {}; v8f acc1 = {};
#pragma unroll
    for (int kc = 0; kc < 4; ++kc) {
      const int koff = kc << 5;
      FragBF A;
      A.q[0] = sq[ln * 16 + ((koff + kA) >> 3)];
      A.q[1] = sq[ln * 16 + ((koff + kA + 8) >> 3)];
      acc0 = wmma_bf16(A.v, Bw[kc][0].v, acc0);
      acc1 = wmma_bf16(A.v, Bw[kc][1].v, acc1);
    }
#pragma unroll
    for (int r = 0; r < 8; ++r) {
      const int jrow = j0 + mb + r;
      const float c = sConf[mb + r];
      const float p0 = acc0[r] + a1c0 + B1c[(size_t)jrow * HID + hb + ln];
      const float p1 = acc1[r] + a1c1 + B1c[(size_t)jrow * HID + hb + 16 + ln];
      g0 += c * fmaxf(p0, 0.f);
      g1 += c * fmaxf(p1, 0.f);
    }
  }
  g0 += __shfl_xor(g0, 16, 32);            // combine the two M halves
  g1 += __shfl_xor(g1, 16, 32);
  if (lane < 16) {
    float* Gp = Gpart + (size_t)seg * NN * HID + (size_t)i * HID;
    Gp[hb + ln]      = g0;
    Gp[hb + 16 + ln] = g1;
  }
}

// ---------------------------------------------------------------------------
// k3: out = (sum_seg Gpart[seg]) @ cw2.T + rowsum * cb2
// ---------------------------------------------------------------------------
__global__ void k3_out(const float* __restrict__ Gpart, const float* __restrict__ cw2,
                       const float* __restrict__ cb2, const float* __restrict__ rowsum,
                       float* __restrict__ out) {
  __shared__ float sG[HID];
  const int i = blockIdx.x, o = threadIdx.x;
  float g = 0.f;
#pragma unroll
  for (int seg = 0; seg < KSPLIT; ++seg)
    g += Gpart[(size_t)seg * NN * HID + (size_t)i * HID + o];
  sG[o] = g;
  __syncthreads();
  float s = cb2[o] * rowsum[i];
  const float* w = cw2 + (size_t)o * HID;
#pragma unroll 4
  for (int h = 0; h < HID; ++h) s += sG[h] * w[h];
  out[(size_t)i * DOUT + o] = s;
}

// ---------------------------------------------------------------------------
extern "C" void kernel_launch(void* const* d_in, const int* in_sizes, int n_in,
                              void* d_out, int out_size, void* d_ws, size_t ws_size,
                              hipStream_t stream) {
  const float* obj   = (const float*)d_in[0];
  const float* bbox  = (const float*)d_in[1];
  const float* mask  = (const float*)d_in[2];
  const float* w1    = (const float*)d_in[3];
  const float* b1    = (const float*)d_in[4];
  const float* gamma = (const float*)d_in[5];
  const float* beta  = (const float*)d_in[6];
  const float* mean  = (const float*)d_in[7];
  const float* var   = (const float*)d_in[8];
  const float* w2    = (const float*)d_in[9];
  const float* b2    = (const float*)d_in[10];
  const float* cw1   = (const float*)d_in[11];
  const float* cb1   = (const float*)d_in[12];
  const float* cw2   = (const float*)d_in[13];
  const float* cb2   = (const float*)d_in[14];
  float* out = (float*)d_out;

  // workspace layout (~5.4 MB total)
  char* ws = (char*)d_ws;
  float* A1    = (float*)(ws + 0);          // 512 KB
  float* B1    = (float*)(ws + 524288);     // 512 KB
  float* A1c   = (float*)(ws + 1048576);    // 512 KB
  float* B1c   = (float*)(ws + 1572864);    // 512 KB
  float* conf  = (float*)(ws + 2097152);    // 1 MB (logits, then conf in place)
  float* Gpart = (float*)(ws + 3145728);    // KSPLIT * 512 KB = 2 MB
  unsigned short* wbf  = (unsigned short*)(ws + 5242880);  // 64 KB
  unsigned short* cwbf = (unsigned short*)(ws + 5308416);  // 64 KB
  float* w2s    = (float*)(ws + 5373952);   // 1 KB
  float* rowsum = (float*)(ws + 5374976);   // 2 KB
  float* Cc     = (float*)(ws + 5377024);   // 4 B

  k_init<<<1, 256, 0, stream>>>(w2, b2, gamma, beta, mean, var, w2s, Cc);
  k_cvt_w<<<256, 256, 0, stream>>>(w1, cw1, wbf, cwbf);
  k0_proj<<<NN, 256, 0, stream>>>(obj, w1, b1, cw1, cb1, A1, B1, A1c, B1c);
  k1_logits<<<NN * (NN / 16), 256, 0, stream>>>(bbox, A1, B1, wbf, w2s, Cc, conf);
  k_softmax<<<NN, 256, 0, stream>>>(conf, mask, rowsum);
  k2_aggregate<<<NN * KSPLIT, 256, 0, stream>>>(bbox, A1c, B1c, cwbf, conf, Gpart);
  k3_out<<<NN, 256, 0, stream>>>(Gpart, cw2, cb2, rowsum, out);
}
